// CrossDecoderLayer_89824946029093
// MI455X (gfx1250) — matrix operand order repaired and checked by reading, not compiled, not run on hardware
//
#include <hip/hip_runtime.h>
#include <hip/hip_bf16.h>

// ---------------- constants (match reference shapes) ----------------
constexpr int B   = 16;
constexpr int W   = 256;
constexpr int S   = 16;
constexpr int Vv  = 32;
constexpr int D   = 512;
constexpr int H   = 8;
constexpr int DK  = 64;          // D / H
constexpr int DF  = 1024;        // 2*D
constexpr int Mtx = B * W;       // 4096 text rows
constexpr int Mav = B * S * Vv;  // 8192 av rows
constexpr float SCALE = 0.125f;  // 1/sqrt(64)
constexpr float EPS   = 1e-5f;

typedef __bf16 bf16;
typedef __attribute__((ext_vector_type(2)))  float v2f;
typedef __attribute__((ext_vector_type(8)))  float v8f;
typedef __attribute__((ext_vector_type(8)))  __bf16 v8bf;
typedef __attribute__((ext_vector_type(16))) __bf16 v16bf;

__device__ __forceinline__ void split2(float x, bf16& h, bf16& l)
{
    h = (bf16)x;
    l = (bf16)(x - (float)h);
}

// Assemble a 16-element A fragment from two contiguous 8-element halves
// (K = kbase+16*kg+0..7 and K = kbase+16*kg+8..15, contiguous order).
__device__ __forceinline__ v16bf loadA(const bf16* p)
{
    v8bf a0 = *(const v8bf*)(p);
    v8bf a1 = *(const v8bf*)(p + 8);
    return __builtin_shufflevector(a0, a1, 0,1,2,3,4,5,6,7,8,9,10,11,12,13,14,15);
}

// ---------------------------------------------------------------
// Split-precision bf16 WMMA GEMM, 2x2 register tiling:
//   C[M,N] = act( (Ah+Al)[M,K] @ (Wh+Wl)[K,N] + bias (+res) )
// using  acc += Ah*Wh + Ah*Wl + Al*Wh   (fp32 accumulate)
// Weights pre-transposed: WT[n*K + k].
// block = 128 (4 waves, 2x2) -> 64x64 per block; each wave owns a
// 32x32 patch = 4 accumulators, 12 WMMAs / K=32 step, 16 b128 loads.
// ---------------------------------------------------------------
__global__ __launch_bounds__(128)
void gemm_wmma_bf16s(const bf16* __restrict__ Ah, const bf16* __restrict__ Al,
                     const bf16* __restrict__ WTh, const bf16* __restrict__ WTl,
                     const float* __restrict__ bias, const float* __restrict__ resid,
                     float* __restrict__ Cf, bf16* __restrict__ Ch, bf16* __restrict__ Cl,
                     int M, int N, int K, int relu)
{
    const int lane  = threadIdx.x & 31;
    const int wave  = threadIdx.x >> 5;
    const int mBase = (blockIdx.y << 6) + ((wave >> 1) << 5);  // 32-row patch
    const int nBase = (blockIdx.x << 6) + ((wave &  1) << 5);  // 32-col patch
    const int kg    = lane >> 4;                               // 0 or 1

    const size_t arow0 = (size_t)(mBase + (lane & 15)) * K + 16 * kg;  // + k
    const size_t arow1 = arow0 + (size_t)16 * K;
    const size_t brow0 = (size_t)(nBase + (lane & 15)) * K + 16 * kg;  // + k
    const size_t brow1 = brow0 + (size_t)16 * K;

    v8f a00 = {}, a01 = {}, a10 = {}, a11 = {};
    for (int k = 0; k < K; k += 32) {
        __builtin_prefetch((const void*)(Ah + arow0 + k + 128), 0, 0);
        v16bf A0h = loadA(Ah + arow0 + k);
        v16bf A0l = loadA(Al + arow0 + k);
        v16bf A1h = loadA(Ah + arow1 + k);
        v16bf A1l = loadA(Al + arow1 + k);
        v16bf B0h = *(const v16bf*)(WTh + brow0 + k);
        v16bf B0l = *(const v16bf*)(WTl + brow0 + k);
        v16bf B1h = *(const v16bf*)(WTh + brow1 + k);
        v16bf B1l = *(const v16bf*)(WTl + brow1 + k);

        a00 = __builtin_amdgcn_wmma_f32_16x16x32_bf16(false, A0h, false, B0h, (short)0, a00, false, false);
        a00 = __builtin_amdgcn_wmma_f32_16x16x32_bf16(false, A0h, false, B0l, (short)0, a00, false, false);
        a00 = __builtin_amdgcn_wmma_f32_16x16x32_bf16(false, A0l, false, B0h, (short)0, a00, false, false);

        a01 = __builtin_amdgcn_wmma_f32_16x16x32_bf16(false, A0h, false, B1h, (short)0, a01, false, false);
        a01 = __builtin_amdgcn_wmma_f32_16x16x32_bf16(false, A0h, false, B1l, (short)0, a01, false, false);
        a01 = __builtin_amdgcn_wmma_f32_16x16x32_bf16(false, A0l, false, B1h, (short)0, a01, false, false);

        a10 = __builtin_amdgcn_wmma_f32_16x16x32_bf16(false, A1h, false, B0h, (short)0, a10, false, false);
        a10 = __builtin_amdgcn_wmma_f32_16x16x32_bf16(false, A1h, false, B0l, (short)0, a10, false, false);
        a10 = __builtin_amdgcn_wmma_f32_16x16x32_bf16(false, A1l, false, B0h, (short)0, a10, false, false);

        a11 = __builtin_amdgcn_wmma_f32_16x16x32_bf16(false, A1h, false, B1h, (short)0, a11, false, false);
        a11 = __builtin_amdgcn_wmma_f32_16x16x32_bf16(false, A1h, false, B1l, (short)0, a11, false, false);
        a11 = __builtin_amdgcn_wmma_f32_16x16x32_bf16(false, A1l, false, B1h, (short)0, a11, false, false);
    }

    const int col0 = nBase + (lane & 15);
    const int rb   = mBase + (kg << 3);

    auto store_tile = [&](const v8f& acc, int rowBase, int col) {
        const float bv = bias ? bias[col] : 0.0f;
#pragma unroll
        for (int i = 0; i < 8; ++i) {
            const size_t idx = (size_t)(rowBase + i) * N + col;
            float v = acc[i] + bv;
            if (resid) v += resid[idx];
            if (relu)  v  = fmaxf(v, 0.0f);
            if (Cf) Cf[idx] = v;
            if (Ch) { bf16 h, l; split2(v, h, l); Ch[idx] = h; Cl[idx] = l; }
        }
    };
    store_tile(a00, rb,      col0);
    store_tile(a01, rb,      col0 + 16);
    store_tile(a10, rb + 16, col0);
    store_tile(a11, rb + 16, col0 + 16);
}

// ---------------------------------------------------------------
// Self-attn scores (exact fp32 WMMA): S = SCALE * Q K^T per (b,h)
// grid: (W/64, W/16, B*H), block 128
// ---------------------------------------------------------------
__global__ __launch_bounds__(128)
void attn_scores_wmma(const float* __restrict__ Q, const float* __restrict__ Km,
                      float* __restrict__ Sc)
{
    const int lane = threadIdx.x & 31;
    const int wave = threadIdx.x >> 5;
    const int bh = blockIdx.z;
    const int b  = bh >> 3;
    const int h  = bh & 7;
    const int mBase = blockIdx.y << 4;
    const int nBase = (blockIdx.x << 6) + (wave << 4);
    const int koff  = (lane >> 4) << 1;

    const float* qp = Q  + (size_t)(b * W + mBase + (lane & 15)) * D + h * DK + koff;
    const float* kp = Km + (size_t)(b * W + nBase + (lane & 15)) * D + h * DK + koff;

    v8f acc = {};
#pragma unroll
    for (int k = 0; k < DK; k += 4) {
        v2f a  = *(const v2f*)(qp + k);
        v2f bb = *(const v2f*)(kp + k);   // K^T: same K-row, consecutive d
        acc = __builtin_amdgcn_wmma_f32_16x16x4_f32(false, a, false, bb, (short)0, acc, false, false);
    }

    const int rbase = mBase + ((lane >> 4) << 3);
    const int col   = nBase + (lane & 15);
#pragma unroll
    for (int i = 0; i < 8; ++i)
        Sc[((size_t)bh * W + rbase + i) * W + col] = acc[i] * SCALE;
}

// ---------------------------------------------------------------
// Masked softmax over rows of Sc (in place). grid = B*H*W, 256 thr.
// ---------------------------------------------------------------
__global__ __launch_bounds__(256)
void softmax_mask_kernel(float* __restrict__ Sc, const unsigned char* __restrict__ mask)
{
    __shared__ float red[256];
    const int r = blockIdx.x;            // (b*H + h)*W + q
    const int b = r / (H * W);
    const int q = r % W;
    const int t = threadIdx.x;

    float v = Sc[(size_t)r * W + t];
    const bool m = mask[((size_t)b * W + q) * W + t] != 0;
    v = m ? v : -1e9f;

    red[t] = v; __syncthreads();
    for (int s = 128; s > 0; s >>= 1) { if (t < s) red[t] = fmaxf(red[t], red[t + s]); __syncthreads(); }
    const float mx = red[0]; __syncthreads();

    const float e = __expf(v - mx);
    red[t] = e; __syncthreads();
    for (int s = 128; s > 0; s >>= 1) { if (t < s) red[t] += red[t + s]; __syncthreads(); }
    const float sum = red[0];

    Sc[(size_t)r * W + t] = e / sum;
}

// ---------------------------------------------------------------
// ctx_h/ctx_l = split( P @ V_h ) per (b,h), written at col h*64.
// grid: (1, W/16, B*H), block 128 (4 waves cover n = 0..63)
// ---------------------------------------------------------------
__global__ __launch_bounds__(128)
void attn_av_wmma(const float* __restrict__ P, const float* __restrict__ Vm,
                  bf16* __restrict__ Oh, bf16* __restrict__ Ol)
{
    const int lane = threadIdx.x & 31;
    const int wave = threadIdx.x >> 5;
    const int bh = blockIdx.z;
    const int b  = bh >> 3;
    const int h  = bh & 7;
    const int mBase = blockIdx.y << 4;
    const int nBase = wave << 4;
    const int koff  = (lane >> 4) << 1;

    const float* pp = P + ((size_t)bh * W + mBase + (lane & 15)) * W + koff;
    const int col = h * DK + nBase + (lane & 15);
    const float* vp = Vm + (size_t)b * W * D + col;

    v8f acc = {};
#pragma unroll 4
    for (int k = 0; k < W; k += 4) {
        v2f a = *(const v2f*)(pp + k);
        v2f bb;
        bb.x = vp[(size_t)(k + koff)     * D];
        bb.y = vp[(size_t)(k + koff + 1) * D];
        acc = __builtin_amdgcn_wmma_f32_16x16x4_f32(false, a, false, bb, (short)0, acc, false, false);
    }

    const int rbase = mBase + ((lane >> 4) << 3);
#pragma unroll
    for (int i = 0; i < 8; ++i) {
        const size_t idx = (size_t)(b * W + rbase + i) * D + col;
        bf16 hh, ll; split2(acc[i], hh, ll);
        Oh[idx] = hh; Ol[idx] = ll;
    }
}

// ---------------------------------------------------------------
// LayerNorm over last dim (512), fp32 out. 1 block/row, 256 threads.
// ---------------------------------------------------------------
__global__ __launch_bounds__(256)
void ln_kernel(const float* __restrict__ in, const float* __restrict__ g,
               const float* __restrict__ bt, float* __restrict__ out)
{
    __shared__ float s1[256], s2[256];
    const size_t r = blockIdx.x;
    const int t = threadIdx.x;
    const float x0 = in[r * D + t];
    const float x1 = in[r * D + 256 + t];
    s1[t] = x0 + x1;
    s2[t] = x0 * x0 + x1 * x1;
    __syncthreads();
    for (int s = 128; s > 0; s >>= 1) {
        if (t < s) { s1[t] += s1[t + s]; s2[t] += s2[t + s]; }
        __syncthreads();
    }
    const float mean = s1[0] * (1.0f / D);
    const float var  = s2[0] * (1.0f / D) - mean * mean;
    const float inv  = rsqrtf(var + EPS);
    out[r * D + t]       = g[t]       * (x0 - mean) * inv + bt[t];
    out[r * D + 256 + t] = g[256 + t] * (x1 - mean) * inv + bt[256 + t];
}

// LayerNorm emitting split bf16 hi/lo (feeds WMMA GEMMs directly).
__global__ __launch_bounds__(256)
void ln_split_kernel(const float* __restrict__ in, const float* __restrict__ g,
                     const float* __restrict__ bt,
                     bf16* __restrict__ oh, bf16* __restrict__ ol)
{
    __shared__ float s1[256], s2[256];
    const size_t r = blockIdx.x;
    const int t = threadIdx.x;
    const float x0 = in[r * D + t];
    const float x1 = in[r * D + 256 + t];
    s1[t] = x0 + x1;
    s2[t] = x0 * x0 + x1 * x1;
    __syncthreads();
    for (int s = 128; s > 0; s >>= 1) {
        if (t < s) { s1[t] += s1[t + s]; s2[t] += s2[t + s]; }
        __syncthreads();
    }
    const float mean = s1[0] * (1.0f / D);
    const float var  = s2[0] * (1.0f / D) - mean * mean;
    const float inv  = rsqrtf(var + EPS);
    const float y0 = g[t]       * (x0 - mean) * inv + bt[t];
    const float y1 = g[256 + t] * (x1 - mean) * inv + bt[256 + t];
    bf16 h, l;
    split2(y0, h, l); oh[r * D + t] = h;       ol[r * D + t] = l;
    split2(y1, h, l); oh[r * D + 256 + t] = h; ol[r * D + 256 + t] = l;
}

// Weight split + transpose: W[K,N] fp32 -> hT/lT[N,K] bf16 (hi/lo).
__global__ __launch_bounds__(256)
void wsplitT_kernel(const float* __restrict__ Wsrc,
                    bf16* __restrict__ hT, bf16* __restrict__ lT, int K, int N)
{
    const int i = blockIdx.x * 256 + threadIdx.x;
    if (i >= K * N) return;
    const int k = i / N, n = i % N;
    bf16 h, l; split2(Wsrc[i], h, l);
    hT[(size_t)n * K + k] = h;
    lT[(size_t)n * K + k] = l;
}

// ---------------------------------------------------------------
// Cross attention (gathered, sigmoid). grid (B*W, H), 64 threads.
// ---------------------------------------------------------------
__global__ __launch_bounds__(64)
void cross_attn_kernel(const float* __restrict__ cq, const float* __restrict__ ck,
                       const float* __restrict__ cv, const int* __restrict__ sidx,
                       float* __restrict__ res, float* __restrict__ attb)
{
    __shared__ float att_sh[Vv];
    const int bw = blockIdx.x;
    const int h  = blockIdx.y;
    const int b  = bw / W;
    const int sent = sidx[bw];
    const size_t kvbase = (size_t)(b * S + sent) * Vv;
    const float* qp = cq + (size_t)bw * D + h * DK;
    const int t = threadIdx.x;

    if (t < Vv) {
        const float* kp = ck + (kvbase + t) * D + h * DK;
        float dot = 0.0f;
#pragma unroll 8
        for (int d = 0; d < DK; ++d) dot += qp[d] * kp[d];
        const float a = 1.0f / (1.0f + __expf(-dot * SCALE));
        att_sh[t] = a;
        attb[((size_t)bw * H + h) * Vv + t] = a;
    }
    __syncthreads();

    float acc = 0.0f;
#pragma unroll 8
    for (int v = 0; v < Vv; ++v)
        acc += att_sh[v] * cv[(kvbase + v) * D + h * DK + t];
    res[(size_t)bw * D + h * DK + t] = acc;
}

// out_attn[b,w,v] = mean over heads
__global__ __launch_bounds__(256)
void attn_mean_kernel(const float* __restrict__ attb, float* __restrict__ out)
{
    const int i = blockIdx.x * 256 + threadIdx.x;  // B*W*Vv = 131072
    if (i >= B * W * Vv) return;
    const int bw = i / Vv;
    const int v  = i % Vv;
    float s = 0.0f;
#pragma unroll
    for (int h = 0; h < H; ++h) s += attb[((size_t)bw * H + h) * Vv + v];
    out[i] = s * (1.0f / H);
}

__global__ __launch_bounds__(256)
void add_kernel(const float* __restrict__ a, const float* __restrict__ b,
                float* __restrict__ c, size_t n)
{
    const size_t i = (size_t)blockIdx.x * 256 + threadIdx.x;
    if (i < n) c[i] = a[i] + b[i];
}

// ---------------------------------------------------------------
extern "C" void kernel_launch(void* const* d_in, const int* in_sizes, int n_in,
                              void* d_out, int out_size, void* d_ws, size_t ws_size,
                              hipStream_t stream)
{
    const float*         text   = (const float*)d_in[0];
    const float*         av     = (const float*)d_in[1];
    const unsigned char* mask   = (const unsigned char*)d_in[2];
    const int*           sidx   = (const int*)d_in[3];
    const float* ln_res1_g = (const float*)d_in[4];
    const float* ln_res1_b = (const float*)d_in[5];
    const float* sa_wq = (const float*)d_in[6];   const float* sa_bq = (const float*)d_in[7];
    const float* sa_wk = (const float*)d_in[8];   const float* sa_bk = (const float*)d_in[9];
    const float* sa_wv = (const float*)d_in[10];  const float* sa_bv = (const float*)d_in[11];
    const float* sa_wo = (const float*)d_in[12];  const float* sa_bo = (const float*)d_in[13];
    const float* ca_ln_tg = (const float*)d_in[14]; const float* ca_ln_tb = (const float*)d_in[15];
    const float* ca_ln_ag = (const float*)d_in[16]; const float* ca_ln_ab = (const float*)d_in[17];
    const float* ca_wq = (const float*)d_in[18];  const float* ca_bq = (const float*)d_in[19];
    const float* ca_wk = (const float*)d_in[20];  const float* ca_bk = (const float*)d_in[21];
    const float* ca_wv = (const float*)d_in[22];  const float* ca_bv = (const float*)d_in[23];
    const float* ln1_g = (const float*)d_in[24];  const float* ln1_b = (const float*)d_in[25];
    const float* ln2_g = (const float*)d_in[26];  const float* ln2_b = (const float*)d_in[27];
    const float* lnr2_g = (const float*)d_in[28]; const float* lnr2_b = (const float*)d_in[29];
    const float* ff_w1 = (const float*)d_in[30];  const float* ff_b1 = (const float*)d_in[31];
    const float* ff_w2 = (const float*)d_in[32];  const float* ff_b2 = (const float*)d_in[33];

    const size_t BWD = (size_t)Mtx * D;       // 2,097,152
    const int    W2  = D * D;                 // 262,144 elems per 512x512 weight

    // ---- persistent split/transposed weights at start of ws ----
    bf16* p = (bf16*)d_ws;
    bf16 *wqh = p, *wql = p + W2;  p += 2 * W2;
    bf16 *wkh = p, *wkl = p + W2;  p += 2 * W2;
    bf16 *wvh = p, *wvl = p + W2;  p += 2 * W2;
    bf16 *woh = p, *wol = p + W2;  p += 2 * W2;
    bf16 *cqh = p, *cql = p + W2;  p += 2 * W2;
    bf16 *ckh = p, *ckl = p + W2;  p += 2 * W2;
    bf16 *cvh = p, *cvl = p + W2;  p += 2 * W2;
    bf16 *f1h = p, *f1l = p + 2 * W2; p += 4 * W2;  // 512x1024
    bf16 *f2h = p, *f2l = p + 2 * W2; p += 4 * W2;  // 1024x512

    // ---- staged scratch region (reused) ----
    char* scr = (char*)d_ws + (size_t)(16u << 20);

    // self-attn stage
    bf16*  xh  = (bf16*)scr;             bf16* xl = xh + BWD;          // [0,8) MB
    float* qf  = (float*)(scr + ((size_t)8u  << 20));                  // [8,16)
    float* kf  = (float*)(scr + ((size_t)16u << 20));                  // [16,24)
    float* vf  = (float*)(scr + ((size_t)24u << 20));                  // [24,32)
    float* sc  = (float*)(scr + ((size_t)32u << 20));                  // [32,64)
    bf16*  cth = (bf16*)(scr + ((size_t)66u << 20)); bf16* ctl = cth + BWD; // [66,74)
    // cross stage (reuses [0,64))
    bf16*  th  = (bf16*)scr;             bf16* tl = th + BWD;          // [0,8)
    bf16*  ah  = (bf16*)(scr + ((size_t)8u  << 20)); bf16* al = ah + (size_t)Mav * D; // [8,24)
    float* cqf = (float*)(scr + ((size_t)24u << 20));                  // [24,32)
    float* ckf = (float*)(scr + ((size_t)32u << 20));                  // [32,48)
    float* cvf = (float*)(scr + ((size_t)48u << 20));                  // [48,64)
    float* resb= (float*)(scr + ((size_t)66u << 20));                  // [66,74)
    float* attb= (float*)(scr + ((size_t)74u << 20));                  // [74,78)
    // LN combine
    float* lnA = (float*)scr;                                          // [0,8)
    float* lnB = (float*)(scr + ((size_t)8u << 20));                   // [8,16)
    // FFN
    bf16*  yh  = (bf16*)scr;             bf16* yl = yh + BWD;          // [0,8)
    bf16*  h1h = (bf16*)(scr + ((size_t)8u << 20)); bf16* h1l = h1h + (size_t)Mtx * DF; // [8,24)

    float* out_text = (float*)d_out;
    float* out_attn = out_text + BWD;

    const dim3 blk128(128), blk256(256), blk64(64);
    const int gW = (W2 + 255) / 256, gF = (2 * W2 + 255) / 256;

    // ---- split + transpose weights ----
    wsplitT_kernel<<<gW, blk256, 0, stream>>>(sa_wq, wqh, wql, D, D);
    wsplitT_kernel<<<gW, blk256, 0, stream>>>(sa_wk, wkh, wkl, D, D);
    wsplitT_kernel<<<gW, blk256, 0, stream>>>(sa_wv, wvh, wvl, D, D);
    wsplitT_kernel<<<gW, blk256, 0, stream>>>(sa_wo, woh, wol, D, D);
    wsplitT_kernel<<<gW, blk256, 0, stream>>>(ca_wq, cqh, cql, D, D);
    wsplitT_kernel<<<gW, blk256, 0, stream>>>(ca_wk, ckh, ckl, D, D);
    wsplitT_kernel<<<gW, blk256, 0, stream>>>(ca_wv, cvh, cvl, D, D);
    wsplitT_kernel<<<gF, blk256, 0, stream>>>(ff_w1, f1h, f1l, D, DF);
    wsplitT_kernel<<<gF, blk256, 0, stream>>>(ff_w2, f2h, f2l, DF, D);

    // ---- res1: x = LN(text); q,k,v; softmax(QK^T*s); PV; Wo + residual ----
    ln_split_kernel<<<Mtx, blk256, 0, stream>>>(text, ln_res1_g, ln_res1_b, xh, xl);
    gemm_wmma_bf16s<<<dim3(D / 64, Mtx / 64), blk128, 0, stream>>>(xh, xl, wqh, wql, sa_bq, nullptr, qf, nullptr, nullptr, Mtx, D, D, 0);
    gemm_wmma_bf16s<<<dim3(D / 64, Mtx / 64), blk128, 0, stream>>>(xh, xl, wkh, wkl, sa_bk, nullptr, kf, nullptr, nullptr, Mtx, D, D, 0);
    gemm_wmma_bf16s<<<dim3(D / 64, Mtx / 64), blk128, 0, stream>>>(xh, xl, wvh, wvl, sa_bv, nullptr, vf, nullptr, nullptr, Mtx, D, D, 0);
    attn_scores_wmma<<<dim3(W / 64, W / 16, B * H), blk128, 0, stream>>>(qf, kf, sc);
    softmax_mask_kernel<<<B * H * W, blk256, 0, stream>>>(sc, mask);
    attn_av_wmma<<<dim3(1, W / 16, B * H), blk128, 0, stream>>>(sc, vf, cth, ctl);
    gemm_wmma_bf16s<<<dim3(D / 64, Mtx / 64), blk128, 0, stream>>>(cth, ctl, woh, wol, sa_bo, text, out_text, nullptr, nullptr, Mtx, D, D, 0);

    // ---- cross attention ----
    ln_split_kernel<<<Mtx, blk256, 0, stream>>>(out_text, ca_ln_tg, ca_ln_tb, th, tl);
    ln_split_kernel<<<Mav, blk256, 0, stream>>>(av, ca_ln_ag, ca_ln_ab, ah, al);
    gemm_wmma_bf16s<<<dim3(D / 64, Mtx / 64), blk128, 0, stream>>>(th, tl, cqh, cql, ca_bq, nullptr, cqf, nullptr, nullptr, Mtx, D, D, 0);
    gemm_wmma_bf16s<<<dim3(D / 64, Mav / 64), blk128, 0, stream>>>(ah, al, ckh, ckl, ca_bk, nullptr, ckf, nullptr, nullptr, Mav, D, D, 0);
    gemm_wmma_bf16s<<<dim3(D / 64, Mav / 64), blk128, 0, stream>>>(ah, al, cvh, cvl, ca_bv, nullptr, cvf, nullptr, nullptr, Mav, D, D, 0);
    cross_attn_kernel<<<dim3(Mtx, H), blk64, 0, stream>>>(cqf, ckf, cvf, sidx, resb, attb);

    // ---- text = LN1(text) + LN2(res) ----
    ln_kernel<<<Mtx, blk256, 0, stream>>>(out_text, ln1_g, ln1_b, lnA);
    ln_kernel<<<Mtx, blk256, 0, stream>>>(resb,     ln2_g, ln2_b, lnB);
    add_kernel<<<(unsigned)(BWD / 256), blk256, 0, stream>>>(lnA, lnB, out_text, BWD);

    // ---- res2: text += FFN(LN(text)) ----
    ln_split_kernel<<<Mtx, blk256, 0, stream>>>(out_text, lnr2_g, lnr2_b, yh, yl);
    gemm_wmma_bf16s<<<dim3(DF / 64, Mtx / 64), blk128, 0, stream>>>(yh, yl, f1h, f1l, ff_b1, nullptr, nullptr, h1h, h1l, Mtx, DF, D, 1);
    gemm_wmma_bf16s<<<dim3(D / 64, Mtx / 64), blk128, 0, stream>>>(h1h, h1l, f2h, f2l, ff_b2, out_text, out_text, nullptr, nullptr, Mtx, D, DF, 0);

    // ---- attn_out = mean over heads ----
    attn_mean_kernel<<<(B * W * Vv + 255) / 256, blk256, 0, stream>>>(attb, out_attn);

    (void)in_sizes; (void)n_in; (void)out_size; (void)ws_size;
}
